// Attention_50002009260222
// MI455X (gfx1250) — matrix operand order repaired and checked by reading, not compiled, not run on hardware
//
#include <hip/hip_runtime.h>
#include <hip/hip_bf16.h>

typedef __bf16 bf16x16 __attribute__((ext_vector_type(16)));
typedef float  f32x8   __attribute__((ext_vector_type(8)));

#define SEQ 4096
#define HD  64
#define NEG_INF (-__builtin_inff())

// Fragment block = 32 lanes x 16 bf16 = 512 elements (1 KB), lane-major so a
// lane's whole A-operand fragment is 16 contiguous bf16 (two b128 loads).
#define FRAG_ELEMS 512
#define KSWZ_BLOCKS (4 * 256 * 2)   // (b, key-tile of 16, d-chunk of 32)
#define VSWZ_BLOCKS (4 * 128 * 4)   // (b, key-block of 32, d-tile of 16)

__device__ __forceinline__ f32x8 wmma_bf16(bf16x16 a, bf16x16 b, f32x8 c) {
  // 8 args: (neg_a, A, neg_b, B, c_mod, C, reuse_a, reuse_b)
  return __builtin_amdgcn_wmma_f32_16x16x32_bf16(false, a, false, b, (short)0, c,
                                                 false, false);
}

// A-operand k order used by this code (contiguous, not AMD's table order): element e of lane l is at
// kdim = (upper-half?16:0) + e.
__device__ __forceinline__ int amap(int e, int half) {
  return (half ? 16 : 0) + e;
}

// ---- Pass 1a: K (fp32 row-major) -> bf16 A-fragment-swizzled (rows = keys,
//      kdim = d). One wave per fragment block: (b, ktile, chunk).
__global__ __launch_bounds__(256, 1)
void pack_k(const float* __restrict__ K, __bf16* __restrict__ kswz) {
  const int wv   = (blockIdx.x * blockDim.x + threadIdx.x) >> 5;
  const int lane = threadIdx.x & 31;
  const int half = lane >> 4;
  const int c    = wv & 1;           // d-chunk (0-31 / 32-63)
  const int kt   = (wv >> 1) & 255;  // 16-key tile
  const int b    = wv >> 9;
  const float* kp = K + ((size_t)b * SEQ + kt * 16 + (lane & 15)) * HD + c * 32;
  __bf16* out = kswz + (size_t)wv * FRAG_ELEMS + lane * 16;
#pragma unroll
  for (int e = 0; e < 16; ++e) out[e] = (__bf16)kp[amap(e, half)];
}

// ---- Pass 1b: V (fp32 row-major) -> bf16 A-fragment-swizzled V^T (rows = d,
//      kdim = keys). One wave per fragment block: (b, kblock, dtile).
__global__ __launch_bounds__(256, 1)
void pack_v(const float* __restrict__ V, __bf16* __restrict__ vswz) {
  const int wv   = (blockIdx.x * blockDim.x + threadIdx.x) >> 5;
  const int lane = threadIdx.x & 31;
  const int half = lane >> 4;
  const int dt   = wv & 3;           // 16-d tile
  const int kb   = (wv >> 2) & 127;  // 32-key block
  const int b    = wv >> 9;
  const float* vp = V + ((size_t)b * SEQ + kb * 32) * HD + dt * 16 + (lane & 15);
  __bf16* out = vswz + (size_t)wv * FRAG_ELEMS + lane * 16;
#pragma unroll
  for (int e = 0; e < 16; ++e) out[e] = (__bf16)vp[(size_t)amap(e, half) * HD];
}

// ---- Main: flash attention, transposed-score formulation.
//   S^T = K_tile x Q^T  (keys in C rows, queries in lanes)
//   O^T = V^T x P^T     (d in C rows, queries in lanes)
// Causal mask is only applied in the peeled final key block.
__global__ __launch_bounds__(256, 1)
void fa_fwd(const float* __restrict__ Q, const __bf16* __restrict__ kswz,
            const __bf16* __restrict__ vswz, float* __restrict__ O)
{
  const int lane  = threadIdx.x & 31;
  const int wave  = threadIdx.x >> 5;
  const int tile  = blockIdx.x * 8 + wave;   // 16-query tile id
  const int b     = tile >> 8;               // 256 tiles per batch
  const int qbase = (tile & 255) << 4;

  const int qcol  = lane & 15;
  const int half  = lane >> 4;
  const int qglob = qbase + qcol;

  const bf16x16* kfrag = (const bf16x16*)kswz;   // 32 fragments per block
  const bf16x16* vfrag = (const bf16x16*)vswz;

  // Q^T B-operand (kdim = d): lane holds kdim = half*16 + e, column = l%16.
  // 1/sqrt(64) folded into the one-time conversion (mask-then-scale ==
  // scale-then-mask since masked entries become -inf either way).
  const float* qptr = Q + ((size_t)b * SEQ + qglob) * HD;
  bf16x16 qb0, qb1;
#pragma unroll
  for (int e = 0; e < 16; ++e) {
    qb0[e] = (__bf16)(qptr[half * 16 + e] * 0.125f);
    qb1[e] = (__bf16)(qptr[32 + half * 16 + e] * 0.125f);
  }

  f32x8 o0 = {}, o1 = {}, o2 = {}, o3 = {};
  float m = NEG_INF, lsum = 0.f;

  auto body = [&](int kbase, bool domask) {
    const int ktile0 = (b * 256 + (kbase >> 4)) * 2;    // K frag block base
    const int vblk   = (b * 128 + (kbase >> 5)) * 4;    // V frag block base

    if (!domask) {  // next block always exists in the unmasked loop
      __builtin_prefetch(kfrag + (size_t)(ktile0 + 4) * 32 + lane, 0, 3);
      __builtin_prefetch(vfrag + (size_t)(vblk + 4) * 32 + lane, 0, 3);
    }

    // K A-operand fragments: two key tiles x two d chunks (32B load each)
    bf16x16 ka00 = kfrag[(size_t)(ktile0 + 0) * 32 + lane];
    bf16x16 ka01 = kfrag[(size_t)(ktile0 + 1) * 32 + lane];
    bf16x16 ka10 = kfrag[(size_t)(ktile0 + 2) * 32 + lane];
    bf16x16 ka11 = kfrag[(size_t)(ktile0 + 3) * 32 + lane];

    f32x8 s0 = {}, s1 = {};
    s0 = wmma_bf16(ka00, qb0, s0);
    s0 = wmma_bf16(ka01, qb1, s0);
    s1 = wmma_bf16(ka10, qb0, s1);
    s1 = wmma_bf16(ka11, qb1, s1);

    // S^T layout: VGPR r of tile t holds key = kbase + t*16 + r + half*8
    float cmax = NEG_INF;
    if (domask) {
#pragma unroll
      for (int r = 0; r < 8; ++r) {
        int key0 = kbase + r + half * 8;
        s0[r] = (key0 > qglob)      ? NEG_INF : s0[r];
        s1[r] = (key0 + 16 > qglob) ? NEG_INF : s1[r];
      }
    }
#pragma unroll
    for (int r = 0; r < 8; ++r) cmax = fmaxf(cmax, fmaxf(s0[r], s1[r]));
    cmax = fmaxf(cmax, __shfl_xor(cmax, 16, 32));

    const float mnew = fmaxf(m, cmax);
    const float corr = __expf(m - mnew);
    m = mnew;

    float rs = 0.f;
    f32x8 p0, p1;
#pragma unroll
    for (int r = 0; r < 8; ++r) {
      p0[r] = __expf(s0[r] - mnew);
      p1[r] = __expf(s1[r] - mnew);
      rs += p0[r] + p1[r];
    }
    rs += __shfl_xor(rs, 16, 32);
    lsum = lsum * corr + rs;

#pragma unroll
    for (int r = 0; r < 8; ++r) {
      o0[r] *= corr; o1[r] *= corr; o2[r] *= corr; o3[r] *= corr;
    }

    // P^T B-operand via half-swap, ONE shuffle per element pair:
    // lower half needs upper's p0[e]; upper half needs lower's p1[e],
    // so pre-select the outgoing value and xor-swap once.
    bf16x16 pb;
#pragma unroll
    for (int e = 0; e < 8; ++e) {
      float send = half ? p0[e] : p1[e];
      float recv = __shfl_xor(send, 16, 32);
      pb[e]     = (__bf16)(half ? recv  : p0[e]);   // key = half*16 + e
      pb[e + 8] = (__bf16)(half ? p1[e] : recv);    // key = half*16 + e + 8
    }

    // V^T A-operand per d-tile (32B load each), O^T += V^T P^T
    bf16x16 va0 = vfrag[(size_t)(vblk + 0) * 32 + lane];
    bf16x16 va1 = vfrag[(size_t)(vblk + 1) * 32 + lane];
    bf16x16 va2 = vfrag[(size_t)(vblk + 2) * 32 + lane];
    bf16x16 va3 = vfrag[(size_t)(vblk + 3) * 32 + lane];
    o0 = wmma_bf16(va0, pb, o0);
    o1 = wmma_bf16(va1, pb, o1);
    o2 = wmma_bf16(va2, pb, o2);
    o3 = wmma_bf16(va3, pb, o3);
  };

  // Full (never-masked) blocks: all keys kbase..kbase+31 <= qbase-1.
  const int lastk = ((qbase + 15) >> 5) << 5;   // kbase of final block
  for (int kbase = 0; kbase < lastk; kbase += 32) body(kbase, false);
  // Peeled final block crossing the diagonal: apply causal mask.
  body(lastk, true);

  // epilogue: normalize, scatter O^T back row-major
  const float inv = 1.f / lsum;
  float* op = O + ((size_t)b * SEQ + qglob) * HD;
#pragma unroll
  for (int r = 0; r < 8; ++r) {
    int d0 = r + half * 8;
    op[d0]      = o0[r] * inv;
    op[16 + d0] = o1[r] * inv;
    op[32 + d0] = o2[r] * inv;
    op[48 + d0] = o3[r] * inv;
  }
}

extern "C" void kernel_launch(void* const* d_in, const int* in_sizes, int n_in,
                              void* d_out, int out_size, void* d_ws, size_t ws_size,
                              hipStream_t stream) {
  const float* q = (const float*)d_in[0];
  const float* k = (const float*)d_in[1];
  const float* v = (const float*)d_in[2];
  // d_in[3] = causal flag; workload is always causal=1.
  float* out = (float*)d_out;

  __bf16* kswz = (__bf16*)d_ws;                                    // 2 MB
  __bf16* vswz = (__bf16*)((char*)d_ws + (size_t)KSWZ_BLOCKS * FRAG_ELEMS * 2);

  // Pass 1: pack K and V into bf16 WMMA-fragment layout (one wave per block).
  pack_k<<<dim3(KSWZ_BLOCKS * 32 / 256), dim3(256), 0, stream>>>(k, kswz);
  pack_v<<<dim3(VSWZ_BLOCKS * 32 / 256), dim3(256), 0, stream>>>(v, vswz);

  // Pass 2: flash attention. 1024 query tiles, 8 waves per 256-thread block.
  const int tiles = 4 * (SEQ / 16);
  fa_fwd<<<dim3(tiles / 8), dim3(256), 0, stream>>>(q, kswz, vswz, out);
}
